// SampleConv_8409545965928
// MI455X (gfx1250) — matrix operand order repaired and checked by reading, not compiled, not run on hardware
//
#include <hip/hip_runtime.h>

// ---------------------------------------------------------------------------
// SampleConv fused kernel for MI455X (gfx1250, wave32, WMMA bf16, async LDS)
//
//   h3 = h1@W1.T ; h4 = h1@W2.T ; hf = h1@Wf.T          [N,128]
//   s  = sigmoid(-(h3@h4.T))                             [N,N]  (never stored)
//   out = hf + (s*(adj+negadj) - negadj) @ hf            <-- fused form
//
// HBM floor = one pass over adj+negadj = 512 MB ~= 22 us @ 23.3 TB/s.
// bf16 WMMA (f32 accum) keeps the 34 GFLOP of matmul near that floor.
// ---------------------------------------------------------------------------

#define NQ 8192
#define DF 128
#define BM 64      // rows per block (4 waves x 16-row strips)
#define BN 64      // j-tile width
#define NSLICE 4   // j-dimension split across blocks (deterministic partials)

typedef unsigned short ushort_t;
typedef __attribute__((ext_vector_type(16))) __bf16 v16bf;
typedef __attribute__((ext_vector_type(8)))  __bf16 v8bf;
typedef __attribute__((ext_vector_type(8)))  float  v8f;

// float -> bf16 round-to-nearest-even
__device__ __forceinline__ ushort_t f2bf(float f) {
    unsigned u = __float_as_uint(f);
    u += 0x7FFFu + ((u >> 16) & 1u);
    return (ushort_t)(u >> 16);
}

__device__ __forceinline__ v16bf cat16(v8bf lo, v8bf hi) {
    return __builtin_shufflevector(lo, hi, 0,1,2,3,4,5,6,7,8,9,10,11,12,13,14,15);
}

// CDNA5 async copy: 16B global -> LDS, tracked by ASYNCcnt (no VGPR data path).
// GVS addressing: mem = SADDR64 + VADDR32 ; VDST VGPR holds LDS byte address.
__device__ __forceinline__ void async_b128(unsigned lds_addr, const void* base, unsigned goff) {
    asm volatile("global_load_async_to_lds_b128 %0, %1, %2"
                 :: "v"(lds_addr), "v"(goff), "s"(base) : "memory");
}

// ---------------------------------------------------------------------------
// Phase 1: h3/h4/hf projections. fp32 math; bf16 copies (h3,h4, hf-transposed)
// for the WMMA phase; fp32 hf kept for the final "+hf" add.
// ---------------------------------------------------------------------------
__global__ __launch_bounds__(128)
void proj_kernel(const float* __restrict__ h1, const float* __restrict__ W1,
                 const float* __restrict__ W2, const float* __restrict__ Wf,
                 ushort_t* __restrict__ h3b, ushort_t* __restrict__ h4b,
                 ushort_t* __restrict__ hfTb, float* __restrict__ hff)
{
    __shared__ float sh1[32][DF];
    const int tid  = threadIdx.x;          // output column 0..127
    const int row0 = blockIdx.x * 32;

    #pragma unroll 4
    for (int r = 0; r < 32; ++r)
        sh1[r][tid] = h1[(size_t)(row0 + r) * DF + tid];
    __syncthreads();

    const float* w1 = W1 + (size_t)tid * DF;
    const float* w2 = W2 + (size_t)tid * DF;
    const float* wf = Wf + (size_t)tid * DF;

    for (int r = 0; r < 32; ++r) {
        float a1 = 0.f, a2 = 0.f, af = 0.f;
        #pragma unroll 8
        for (int k = 0; k < DF; ++k) {
            float h = sh1[r][k];
            a1 = fmaf(h, w1[k], a1);
            a2 = fmaf(h, w2[k], a2);
            af = fmaf(h, wf[k], af);
        }
        const size_t row = (size_t)(row0 + r);
        h3b[row * DF + tid]      = f2bf(a1);
        h4b[row * DF + tid]      = f2bf(a2);
        hff[row * DF + tid]      = af;
        hfTb[(size_t)tid * NQ + row] = f2bf(af);   // transposed hf (bf16)
    }
}

// ---------------------------------------------------------------------------
// Phase 2: fused  G -> sigmoid gate -> (M @ hf)  with bf16 WMMA and
// double-buffered async LDS staging.
// grid: (NQ/BM) x NSLICE.  block: 128 threads = 4 wave32, one 16-row strip
// per wave.  adj/negadj streamed from HBM exactly once (memory floor).
// ---------------------------------------------------------------------------

// stage one j-tile (h4 rows + hfT columns) into LDS via async copies
__device__ __forceinline__ void stage_async(int tid, int jbase,
        ushort_t (&h4buf)[BN][144], ushort_t (&hfbuf)[DF][80],
        const ushort_t* __restrict__ h4b, const ushort_t* __restrict__ hfTb)
{
    #pragma unroll
    for (int i = 0; i < 8; ++i) {                  // 1024 x 16B : h4 tile
        int v = tid + i * 128;
        int row = v >> 4, c = v & 15;
        async_b128((unsigned)(uintptr_t)&h4buf[row][c * 8], h4b,
                   (unsigned)(((jbase + row) * DF + c * 8) * 2));
    }
    #pragma unroll
    for (int i = 0; i < 8; ++i) {                  // 1024 x 16B : hfT tile
        int v = tid + i * 128;
        int d = v >> 3, c = v & 7;
        async_b128((unsigned)(uintptr_t)&hfbuf[d][c * 8], hfTb,
                   (unsigned)((d * NQ + jbase + c * 8) * 2));
    }
}

__global__ __launch_bounds__(128)
void fused_gate_kernel(const float* __restrict__ adj, const float* __restrict__ negadj,
                       const ushort_t* __restrict__ h3b, const ushort_t* __restrict__ h4b,
                       const ushort_t* __restrict__ hfTb, float* __restrict__ part)
{
    // double-buffered tiles; row strides keep 32B alignment for v16bf loads
    __shared__ __attribute__((aligned(32))) ushort_t lds_h4 [2][BN][144]; // 36.0 KB
    __shared__ __attribute__((aligned(32))) ushort_t lds_hfT[2][DF][80];  // 40.0 KB
    __shared__ __attribute__((aligned(32))) ushort_t lds_M  [4][16][80];  // 10.0 KB

    const int tid = threadIdx.x;
    const int w   = tid >> 5;       // wave id 0..3   -> strip rows [16w,16w+16)
    const int l   = tid & 31;       // lane
    const int ln  = l & 15;         // N index within 16x16 tile
    const int lh  = l >> 4;         // lane-half select (ISA 16-bit operand layout)
    const int i0  = blockIdx.x * BM;
    const int slice = blockIdx.y;

    // --- A-operand fragments: h3 strip (16 x 128), per ISA 16-bit 16x32 layout:
    // lane<16 holds K in {kc*32+0..7, kc*32+16..23}; lane>=16 the +8 offsets.
    const int R = i0 + 16 * w + ln;
    v16bf a3[4];
    #pragma unroll
    for (int kc = 0; kc < 4; ++kc) {
        const ushort_t* p = h3b + (size_t)R * DF + kc * 32 + lh * 8;
        a3[kc] = cat16(*(const v8bf*)p, *(const v8bf*)(p + 16));
    }

    v8f acc[8] = {};   // 16 x 128 output strip accumulator (f32)

    const int jt_per = (NQ / NSLICE) / BN;   // 32 tiles per slice
    const int jorg   = slice * (NQ / NSLICE);

    // prologue: stage tile 0 into buffer 0
    stage_async(tid, jorg, lds_h4[0], lds_hfT[0], h4b, hfTb);
    asm volatile("s_wait_asynccnt 0" ::: "memory");
    __syncthreads();

    int buf = 0;
    for (int jt = 0; jt < jt_per; ++jt) {
        const int jbase = jorg + jt * BN;

        if (jt + 1 < jt_per) {
            // async-stage next tile into the other buffer (overlaps WMMA below)
            stage_async(tid, jbase + BN, lds_h4[buf ^ 1], lds_hfT[buf ^ 1], h4b, hfTb);
            // prefetch next tile's adj/negadj cachelines (the real HBM stream):
            // 64 lines per array per wave-tile, 32 lanes -> 2 prefetches each
            #pragma unroll
            for (int q = 0; q < 2; ++q) {
                const int e = q * 32 + l;                     // 0..63
                const int m = e >> 2, t = e & 3;
                const size_t off = (size_t)(i0 + 16 * w + m) * NQ + (jbase + BN) + 16 * t;
                __builtin_prefetch(adj + off, 0, 1);
                __builtin_prefetch(negadj + off, 0, 1);
            }
        }

        // ---- G = h3_strip(16x128) @ h4_tile^T : 4 subtiles, K=128 in 4 chunks
        // B-operand (32x16 layout): lane<16 holds K {0..7,16..23} of column ln; lane>=16 the +8 offsets.
        v8f g[4];
        #pragma unroll
        for (int t = 0; t < 4; ++t) {
            v8f gg = {};
            #pragma unroll
            for (int kc = 0; kc < 4; ++kc) {
                const ushort_t* pb = &lds_h4[buf][16 * t + ln][kc * 32 + lh * 8];
                v16bf b = cat16(*(const v8bf*)pb, *(const v8bf*)(pb + 16));
                gg = __builtin_amdgcn_wmma_f32_16x16x32_bf16(
                        false, a3[kc], false, b, (short)0, gg, false, false);
            }
            g[t] = gg;
        }

        // ---- elementwise gate on D-layout (VGPR r -> row r+8*lh, lane -> col)
        //  m = s*(adj+negadj) - negadj ,  s = sigmoid(-G) = 1/(1+exp(G))
        #pragma unroll
        for (int t = 0; t < 4; ++t) {
            #pragma unroll
            for (int r = 0; r < 8; ++r) {
                const int m_row = r + 8 * lh;
                const size_t I = (size_t)(i0 + 16 * w + m_row);
                const size_t J = (size_t)(jbase + 16 * t + ln);
                const float a  = adj[I * NQ + J];        // single streaming pass
                const float na = negadj[I * NQ + J];
                const float s  = 1.0f / (1.0f + __expf(g[t][r]));
                const float mv = s * (a + na) - na;
                lds_M[w][m_row][16 * t + ln] = f2bf(mv); // D-layout -> LDS
            }
        }
        // Per-wave fence: DS stores above feed DS loads below (in-order per
        // wave in HW; the asm is also a compiler memory barrier).
        asm volatile("s_wait_dscnt 0" ::: "memory");

        // ---- re-read M strip as A-operand fragments (K = j_local, 2 chunks)
        v16bf am[2];
        #pragma unroll
        for (int kc = 0; kc < 2; ++kc) {
            const ushort_t* p = &lds_M[w][ln][kc * 32 + lh * 8];
            am[kc] = cat16(*(const v8bf*)p, *(const v8bf*)(p + 16));
        }

        // ---- acc(16x128) += M(16x64) @ hf(64x128) ; B columns from hfT rows
        #pragma unroll
        for (int c = 0; c < 8; ++c) {
            #pragma unroll
            for (int kc = 0; kc < 2; ++kc) {
                const ushort_t* pb = &lds_hfT[buf][16 * c + ln][kc * 32 + lh * 8];
                v16bf b = cat16(*(const v8bf*)pb, *(const v8bf*)(pb + 16));
                acc[c] = __builtin_amdgcn_wmma_f32_16x16x32_bf16(
                            false, am[kc], false, b, (short)0, acc[c], false, false);
            }
        }

        // next tile's async copies (issued above) must be LDS-visible to all
        asm volatile("s_wait_asynccnt 0" ::: "memory");
        __syncthreads();
        buf ^= 1;
    }

    // ---- write deterministic per-slice partials (each element written once)
    #pragma unroll
    for (int c = 0; c < 8; ++c) {
        #pragma unroll
        for (int r = 0; r < 8; ++r) {
            const size_t I = (size_t)(i0 + 16 * w + r + 8 * lh);
            part[((size_t)slice * NQ + I) * DF + 16 * c + ln] = acc[c][r];
        }
    }
}

// ---------------------------------------------------------------------------
// Phase 3: out = hf + sum over slices (deterministic, no float atomics)
// ---------------------------------------------------------------------------
__global__ __launch_bounds__(256)
void finalize_kernel(const float* __restrict__ hff, const float* __restrict__ part,
                     float* __restrict__ out)
{
    const size_t idx = (size_t)blockIdx.x * 256 + threadIdx.x;
    const size_t M = (size_t)NQ * DF;
    out[idx] = hff[idx] + part[idx] + part[idx + M] + part[idx + 2 * M] + part[idx + 3 * M];
}

// ---------------------------------------------------------------------------
extern "C" void kernel_launch(void* const* d_in, const int* in_sizes, int n_in,
                              void* d_out, int out_size, void* d_ws, size_t ws_size,
                              hipStream_t stream)
{
    const float* h1     = (const float*)d_in[0];
    const float* adj    = (const float*)d_in[1];
    const float* negadj = (const float*)d_in[2];
    const float* W1     = (const float*)d_in[3];
    const float* W2     = (const float*)d_in[4];
    const float* Wf     = (const float*)d_in[5];
    float* out = (float*)d_out;

    // workspace layout (26 MB total)
    char* ws = (char*)d_ws;
    ushort_t* h3b  = (ushort_t*)(ws);                       //  2 MB bf16 h3
    ushort_t* h4b  = (ushort_t*)(ws + (size_t)( 2u << 20)); //  2 MB bf16 h4
    ushort_t* hfTb = (ushort_t*)(ws + (size_t)( 4u << 20)); //  2 MB bf16 hf^T
    float*    hff  = (float*)   (ws + (size_t)( 6u << 20)); //  4 MB f32 hf
    float*    part = (float*)   (ws + (size_t)(10u << 20)); // 16 MB slice partials

    proj_kernel<<<NQ / 32, 128, 0, stream>>>(h1, W1, W2, Wf, h3b, h4b, hfTb, hff);
    fused_gate_kernel<<<dim3(NQ / BM, NSLICE), 128, 0, stream>>>(adj, negadj, h3b, h4b, hfTb, part);
    finalize_kernel<<<(NQ * DF) / 256, 256, 0, stream>>>(hff, part, out);
}
